// Model2b_47605417509341
// MI455X (gfx1250) — matrix operand order repaired and checked by reading, not compiled, not run on hardware
//
#include <hip/hip_runtime.h>

#define DEVI __device__ __forceinline__

typedef __attribute__((ext_vector_type(16))) _Float16 v16h;
typedef __attribute__((ext_vector_type(8)))  _Float16 v8h;
typedef __attribute__((ext_vector_type(8)))  float    v8f;
typedef __attribute__((ext_vector_type(4)))  unsigned int u32x4;
typedef __attribute__((ext_vector_type(8)))  int i32x8;
typedef __attribute__((ext_vector_type(4)))  int i32x4;

union V16H { v16h v; v8h h8[2]; };

DEVI v8f zf8() { v8f z; for (int i = 0; i < 8; ++i) z[i] = 0.0f; return z; }
DEVI v8h zh8() { v8h z; for (int i = 0; i < 8; ++i) z[i] = (_Float16)0.0f; return z; }

DEVI v8f wmma16(v16h a, v16h b, v8f c) {
  return __builtin_amdgcn_wmma_f32_16x16x32_f16(false, a, false, b, (short)0, c,
                                                false, false);
}

// Load a 16x32 f16 A/B fragment as two 16-byte chunks.
DEVI v16h ldA(const _Float16* p0, const _Float16* p1) {
  V16H r;
  r.h8[0] = *(const v8h*)p0;
  r.h8[1] = *(const v8h*)p1;
  return r.v;
}

// --- CDNA5 async global->LDS (VGLOBAL GVS mode: SGPR base + VGPR offset) ----
DEVI void async_ld_b128(unsigned lds_addr, unsigned long long gbase, unsigned voff) {
  asm volatile("global_load_async_to_lds_b128 %0, %1, %2"
               :: "v"(lds_addr), "v"(voff), "s"(gbase) : "memory");
}
DEVI void wait_async0() { asm volatile("s_wait_asynccnt 0x0" ::: "memory"); }
DEVI void wait_async1() { asm volatile("s_wait_asynccnt 0x1" ::: "memory"); }
DEVI unsigned lds_off(const void* p) { return (unsigned)(size_t)p; }

// ---------------------------------------------------------------------------
// Small utility kernels
// ---------------------------------------------------------------------------
__global__ void zero_i32(int* p, int n) {
  int i = blockIdx.x * 256 + threadIdx.x;
  if (i < n) p[i] = 0;
}

// f32 [K][N] row-major  ->  f16 [Npad][K] (W^T, zero-padded rows)
__global__ void convT_kernel(const float* __restrict__ in, _Float16* __restrict__ out,
                             int K, int N, int Npad) {
  int idx = blockIdx.x * 256 + threadIdx.x;
  if (idx >= Npad * K) return;
  int n = idx / K, k = idx - n * K;
  out[idx] = (n < N) ? (_Float16)in[(size_t)k * N + n] : (_Float16)0.0f;
}

// tcn weight f32 [O=128][I=128][K=3] -> f16 [k][o][i]
__global__ void convTcn_kernel(const float* __restrict__ in, _Float16* __restrict__ out) {
  int idx = blockIdx.x * 256 + threadIdx.x;
  if (idx >= 3 * 128 * 128) return;
  int k = idx / (128 * 128);
  int o = (idx >> 7) & 127;
  int i = idx & 127;
  out[idx] = (_Float16)in[((size_t)o * 128 + i) * 3 + k];
}

// ---------------------------------------------------------------------------
// Embedding + active-agent count.  h16[bn][t][e] = x@w_emb + b_emb  (f16)
// ---------------------------------------------------------------------------
__global__ __launch_bounds__(256) void embed_kernel(
    const float* __restrict__ x, const float* __restrict__ w_emb,
    const float* __restrict__ b_emb, _Float16* __restrict__ h16,
    int* __restrict__ nb_active) {
  const int bn = blockIdx.x;
  const float* xa = x + (size_t)bn * 32;                // [16][2]
  if (threadIdx.x == 0) {
    float s = 0.0f;
    for (int i = 0; i < 32; ++i) s += xa[i];
    if (s > 0.0f) atomicAdd(&nb_active[bn >> 8], 1);
  }
  for (int u = threadIdx.x; u < 2048; u += 256) {
    const int t = u >> 7, e = u & 127;
    float vv = xa[2 * t] * w_emb[e] + xa[2 * t + 1] * w_emb[128 + e] + b_emb[e];
    h16[(size_t)bn * 2048 + u] = (_Float16)vv;
  }
}

// Zero TCN output rows of inactive agents (mask = n < nb_active[b])
__global__ __launch_bounds__(256) void mask_kernel(_Float16* __restrict__ h16,
                                                   const int* __restrict__ nb) {
  const int bn = blockIdx.x;
  if ((bn & 255) < nb[bn >> 8]) return;
  *(v8h*)&h16[(size_t)bn * 2048 + threadIdx.x * 8] = zh8();
}

// ---------------------------------------------------------------------------
// Dilated causal conv as WMMA GEMM.
// in/out: f16 [BN][T=16][E=128].  wT: f16 [tap k][cout][cin].
// Weight panel (96 KB) is fetched by the Tensor Data Mover; per-agent input
// rows arrive via async global->LDS loads.
// ---------------------------------------------------------------------------
__global__ __launch_bounds__(256) void tcn_conv_kernel(
    const _Float16* __restrict__ in, const _Float16* __restrict__ wT,
    const float* __restrict__ bias, _Float16* __restrict__ out,
    const _Float16* __restrict__ resid, int dil) {
  __shared__ _Float16 Wt[3 * 128 * 128];   // 96 KB, filled by TDM
  __shared__ _Float16 Xp[8][24 * 128];     // 8 zero halo rows + 16 time rows
  const int tid = threadIdx.x, wave = tid >> 5, lane = tid & 31;
  const int m = lane & 15, half = lane >> 4;
  const int agent = blockIdx.x * 8 + wave;

  if (tid < 32) {
    // D# for a 1-row 2-D tile: 12288 x 8-byte elements == whole weight panel.
    unsigned long long ga = (unsigned long long)(size_t)wT;
    u32x4 g0;
    g0[0] = 1u;                                    // count = 1 valid descriptor
    g0[1] = lds_off(&Wt[0]);                       // LDS destination
    g0[2] = (unsigned)(ga & 0xffffffffu);          // global_addr[31:0]
    g0[3] = (unsigned)((ga >> 32) & 0x01ffffffu) | (2u << 30);  // addr hi | type=2
    i32x8 g1;
    g1[0] = (int)(3u << 16);                       // workgroup_mask=0, data_size=8B
    g1[1] = (int)((12288u & 0xffffu) << 16);       // tensor_dim0 lo16
    g1[2] = (int)((12288u >> 16) | (1u << 16));    // tensor_dim0 hi | tensor_dim1=1
    g1[3] = (int)(12288u << 16);                   // tile_dim0 = 12288
    g1[4] = 1;                                     // tile_dim1 = 1
    g1[5] = 12288;                                 // tensor_dim0_stride lo
    g1[6] = 0;
    g1[7] = 0;
    i32x4 z4; z4[0] = 0; z4[1] = 0; z4[2] = 0; z4[3] = 0;
    i32x8 z8; for (int i = 0; i < 8; ++i) z8[i] = 0;
    __builtin_amdgcn_tensor_load_to_lds(g0, g1, z4, z4, z8, 0);
  }

  {
    v8h z = zh8();
    for (int u = lane * 8; u < 8 * 128; u += 32 * 8) *(v8h*)&Xp[wave][u] = z;
    const unsigned abase = (unsigned)agent * 4096u;  // byte offset of this agent
    for (int u = lane * 8; u < 2048; u += 32 * 8)
      async_ld_b128(lds_off(&Xp[wave][8 * 128 + u]),
                    (unsigned long long)(size_t)in, abase + (unsigned)(u * 2));
  }
  wait_async0();
  if (tid < 32) __builtin_amdgcn_s_wait_tensorcnt(0);
  __syncthreads();

  v8f acc[8];
  for (int i = 0; i < 8; ++i) acc[i] = zf8();
#pragma unroll
  for (int k = 0; k < 3; ++k) {
    const _Float16* arow = &Xp[wave][(8 + m - (2 - k) * dil) * 128];
    const _Float16* wk = &Wt[k * 128 * 128];
#pragma unroll
    for (int c = 0; c < 4; ++c) {  // cin chunks of 32
      v16h a = ldA(arow + c * 32 + 16 * half, arow + c * 32 + 8 + 16 * half);
#pragma unroll
      for (int nt = 0; nt < 8; ++nt) {  // cout tiles of 16
        const _Float16* bc = &wk[(nt * 16 + m) * 128 + c * 32 + 16 * half];
        acc[nt] = wmma16(a, ldA(bc, bc + 8), acc[nt]);
      }
    }
  }
  const size_t base = (size_t)agent * 2048;
#pragma unroll
  for (int nt = 0; nt < 8; ++nt) {
    const int col = nt * 16 + m;
    const float bv = bias[col];
#pragma unroll
    for (int j = 0; j < 8; ++j) {
      const int row = j + 8 * half;
      float vv = fmaxf(acc[nt][j] + bv, 0.0f);
      if (resid) vv = fmaxf((float)resid[base + row * 128 + col] + vv, 0.0f);
      out[base + row * 128 + col] = (_Float16)vv;
    }
  }
}

// ---------------------------------------------------------------------------
// Generic GEMM: out[M,N] = A[M,K] @ Bt[N,K]^T + bias, optional relu.
// B panel double-buffered in LDS via async global->LDS loads (ASYNCcnt),
// issued one k-step ahead of the WMMA consumer.
// ---------------------------------------------------------------------------
__global__ __launch_bounds__(256) void gemm_kernel(
    const _Float16* __restrict__ A, const _Float16* __restrict__ Bt,
    const float* __restrict__ bias, _Float16* __restrict__ outH,
    float* __restrict__ outF, int M, int N, int K, int relu) {
  __shared__ _Float16 Bs[2][64 * 32];
  const int tid = threadIdx.x, wave = tid >> 5, lane = tid & 31;
  const int m = lane & 15, half = lane >> 4;
  const int mbase = blockIdx.x * 128;
  const int nbase = blockIdx.y * 64;
  const _Float16* arow = A + (size_t)(mbase + wave * 16 + m) * K;

  const int u = tid * 8;
  const int nl = u >> 5, kk = u & 31;
  const unsigned long long btb = (unsigned long long)(size_t)Bt;
  const unsigned grow = (unsigned)(((size_t)(nbase + nl) * K + kk) * 2);
  const unsigned ldsB[2] = {lds_off(&Bs[0][u]), lds_off(&Bs[1][u])};

  v8f acc[4];
  for (int i = 0; i < 4; ++i) acc[i] = zf8();

  const int steps = K >> 5;
  async_ld_b128(ldsB[0], btb, grow);
  for (int it = 0; it < steps; ++it) {
    const int cur = it & 1;
    if (it + 1 < steps) {
      async_ld_b128(ldsB[cur ^ 1], btb, grow + (unsigned)((it + 1) << 6));
      wait_async1();               // oldest (current buffer) has landed
    } else {
      wait_async0();
    }
    __syncthreads();               // all waves' panel pieces visible
    const int k0 = it << 5;
    __builtin_prefetch(arow + k0 + 128, 0, 1);
    v16h a = ldA(arow + k0 + 16 * half, arow + k0 + 8 + 16 * half);
#pragma unroll
    for (int nt = 0; nt < 4; ++nt) {
      const _Float16* bc = &Bs[cur][(nt * 16 + m) * 32 + 16 * half];
      acc[nt] = wmma16(a, ldA(bc, bc + 8), acc[nt]);
    }
    __syncthreads();               // reads done before next overwrite
  }
#pragma unroll
  for (int nt = 0; nt < 4; ++nt) {
    const int col = nbase + nt * 16 + m;
    if (col >= N) continue;
    const float bv = bias[col];
#pragma unroll
    for (int j = 0; j < 8; ++j) {
      const int row = mbase + wave * 16 + j + 8 * half;
      float vv = acc[nt][j] + bv;
      if (relu) vv = fmaxf(vv, 0.0f);
      if (outH) outH[(size_t)row * N + col] = (_Float16)vv;
      else      outF[(size_t)row * N + col] = vv;
    }
  }
}

// ---------------------------------------------------------------------------
// Multi-head self-attention over agents.  One block per (b,h): 256 agents,
// dk=dv=64.  Scores kept in accumulator VGPRs (16 x v8f / wave), softmax via
// shfl_xor within the 16-lane C-layout groups, P staged to LDS for P@V.
// K panel arrives via async global->LDS.
// ---------------------------------------------------------------------------
__global__ __launch_bounds__(256) void attn_kernel(
    const _Float16* __restrict__ q, const _Float16* __restrict__ k,
    const _Float16* __restrict__ v, _Float16* __restrict__ o) {
  __shared__ _Float16 Ks[256 * 64];     // [agent][d]   32 KB
  __shared__ _Float16 VT[64 * 256];     // [d][agent]   32 KB
  __shared__ _Float16 P[8][16 * 256];   // per-wave probs, 64 KB
  const int bh = blockIdx.x, b = bh >> 3, h = bh & 7;
  const int tid = threadIdx.x, wave = tid >> 5, lane = tid & 31;
  const int m = lane & 15, half = lane >> 4;
  const size_t hoff = (size_t)h * 64;

  for (int u = tid * 8; u < 256 * 64; u += 256 * 8) {
    int n = u >> 6, d = u & 63;
    async_ld_b128(lds_off(&Ks[u]), (unsigned long long)(size_t)k,
                  (unsigned)((((size_t)(b * 256 + n)) * 512 + hoff + d) * 2));
  }
  for (int u = tid; u < 256 * 64; u += 256) {
    int d = u >> 8, n = u & 255;
    VT[u] = v[((size_t)(b * 256 + n)) * 512 + hoff + d];
  }
  wait_async0();
  __syncthreads();

  for (int qi = 0; qi < 2; ++qi) {
    const int m0 = (wave * 2 + qi) * 16;
    const _Float16* qrow = q + ((size_t)(b * 256 + m0 + m)) * 512 + hoff;
    v16h a0 = ldA(qrow + 16 * half, qrow + 8 + 16 * half);
    v16h a1 = ldA(qrow + 32 + 16 * half, qrow + 40 + 16 * half);

    v8f s[16];
#pragma unroll
    for (int kt = 0; kt < 16; ++kt) {
      s[kt] = zf8();
      const _Float16* kc = &Ks[(kt * 16 + m) * 64];
      s[kt] = wmma16(a0, ldA(kc + 16 * half, kc + 8 + 16 * half), s[kt]);
      s[kt] = wmma16(a1, ldA(kc + 32 + 16 * half, kc + 40 + 16 * half), s[kt]);
    }
    const float scale = 0.125f;  // 1/sqrt(64)
#pragma unroll
    for (int j = 0; j < 8; ++j) {
      float mx = -3.0e38f;
#pragma unroll
      for (int kt = 0; kt < 16; ++kt) mx = fmaxf(mx, s[kt][j] * scale);
      for (int dd = 1; dd < 16; dd <<= 1) mx = fmaxf(mx, __shfl_xor(mx, dd, 32));
      float sum = 0.0f;
#pragma unroll
      for (int kt = 0; kt < 16; ++kt) {
        float e = __expf(s[kt][j] * scale - mx);
        s[kt][j] = e;
        sum += e;
      }
      for (int dd = 1; dd < 16; dd <<= 1) sum += __shfl_xor(sum, dd, 32);
      const float inv = 1.0f / sum;
      const int row = j + 8 * half;
#pragma unroll
      for (int kt = 0; kt < 16; ++kt)
        P[wave][row * 256 + kt * 16 + m] = (_Float16)(s[kt][j] * inv);
    }
#pragma unroll
    for (int ot = 0; ot < 4; ++ot) {   // v-dim tiles of 16
      v8f oa = zf8();
#pragma unroll
      for (int mc = 0; mc < 8; ++mc) { // agent chunks of 32
        const _Float16* pr = &P[wave][m * 256 + mc * 32];
        const _Float16* vc = &VT[(ot * 16 + m) * 256 + mc * 32 + 16 * half];
        oa = wmma16(ldA(pr + 16 * half, pr + 8 + 16 * half), ldA(vc, vc + 8), oa);
      }
#pragma unroll
      for (int j = 0; j < 8; ++j)
        o[((size_t)(b * 256 + m0 + j + 8 * half)) * 512 + hoff + ot * 16 + m] =
            (_Float16)oa[j];
    }
  }
}

// ---------------------------------------------------------------------------
extern "C" void kernel_launch(void* const* d_in, const int* in_sizes, int n_in,
                              void* d_out, int out_size, void* d_ws, size_t ws_size,
                              hipStream_t stream) {
  (void)in_sizes; (void)n_in; (void)out_size; (void)ws_size;
  const float* x      = (const float*)d_in[0];
  // d_in[1] = types (unused by reference math)
  const float* w_emb  = (const float*)d_in[2];
  const float* b_emb  = (const float*)d_in[3];
  const float* tcn_w1 = (const float*)d_in[4];
  const float* tcn_b1 = (const float*)d_in[5];
  const float* tcn_w2 = (const float*)d_in[6];
  const float* tcn_b2 = (const float*)d_in[7];
  const float* w_enc  = (const float*)d_in[8];
  const float* b_enc  = (const float*)d_in[9];
  const float* wq = (const float*)d_in[10]; const float* bq = (const float*)d_in[11];
  const float* wk = (const float*)d_in[12]; const float* bk = (const float*)d_in[13];
  const float* wv = (const float*)d_in[14]; const float* bv = (const float*)d_in[15];
  const float* wo = (const float*)d_in[16]; const float* bo = (const float*)d_in[17];
  const float* w_p1 = (const float*)d_in[18]; const float* b_p1 = (const float*)d_in[19];
  const float* w_p2 = (const float*)d_in[20]; const float* b_p2 = (const float*)d_in[21];
  const float* w_p3 = (const float*)d_in[22]; const float* b_p3 = (const float*)d_in[23];

  char* ws = (char*)d_ws;
  size_t off = 0;
  auto alloc = [&](size_t bytes) {
    size_t o = off;
    off += (bytes + 255) & ~(size_t)255;
    return o;
  };
  const size_t BN = 16384;
  _Float16* h16   = (_Float16*)(ws + alloc(BN * 2048 * 2));
  _Float16* z16   = (_Float16*)(ws + alloc(BN * 2048 * 2));
  _Float16* feat  = (_Float16*)(ws + alloc(BN * 512 * 2));
  _Float16* qb    = (_Float16*)(ws + alloc(BN * 512 * 2));
  _Float16* kb    = (_Float16*)(ws + alloc(BN * 512 * 2));
  _Float16* vb    = (_Float16*)(ws + alloc(BN * 512 * 2));
  _Float16* ob    = (_Float16*)(ws + alloc(BN * 512 * 2));
  _Float16* projb = (_Float16*)(ws + alloc(BN * 512 * 2));
  _Float16* wencT = (_Float16*)(ws + alloc((size_t)512 * 2048 * 2));
  _Float16* wqT   = (_Float16*)(ws + alloc((size_t)512 * 512 * 2));
  _Float16* wkT   = (_Float16*)(ws + alloc((size_t)512 * 512 * 2));
  _Float16* wvT   = (_Float16*)(ws + alloc((size_t)512 * 512 * 2));
  _Float16* woT   = (_Float16*)(ws + alloc((size_t)512 * 512 * 2));
  _Float16* wp1T  = (_Float16*)(ws + alloc((size_t)1024 * 512 * 2));
  _Float16* wp2T  = (_Float16*)(ws + alloc((size_t)1024 * 1024 * 2));
  _Float16* wp3T  = (_Float16*)(ws + alloc((size_t)64 * 1024 * 2));  // N=24 padded to 64
  _Float16* tcnT[6];
  for (int i = 0; i < 6; ++i) tcnT[i] = (_Float16*)(ws + alloc((size_t)3 * 128 * 128 * 2));
  int* nbact = (int*)(ws + alloc(64 * 4));
  _Float16* p1b = h16;  // dead after encoder GEMM -> reuse for MLP layer 1
  _Float16* p2b = z16;  // dead after TCN          -> reuse for MLP layer 2

  dim3 blk(256);
  zero_i32<<<1, 64, 0, stream>>>(nbact, 64);
  embed_kernel<<<dim3((unsigned)BN), blk, 0, stream>>>(x, w_emb, b_emb, h16, nbact);

  auto launch_convT = [&](const float* in, _Float16* outp, int K, int N, int Npad) {
    int tot = Npad * K;
    convT_kernel<<<dim3((tot + 255) / 256), blk, 0, stream>>>(in, outp, K, N, Npad);
  };
  launch_convT(w_enc, wencT, 2048, 512, 512);
  launch_convT(wq, wqT, 512, 512, 512);
  launch_convT(wk, wkT, 512, 512, 512);
  launch_convT(wv, wvT, 512, 512, 512);
  launch_convT(wo, woT, 512, 512, 512);
  launch_convT(w_p1, wp1T, 512, 1024, 1024);
  launch_convT(w_p2, wp2T, 1024, 1024, 1024);
  launch_convT(w_p3, wp3T, 1024, 24, 64);
  for (int i = 0; i < 3; ++i) {
    convTcn_kernel<<<dim3(192), blk, 0, stream>>>(tcn_w1 + (size_t)i * 49152, tcnT[2 * i]);
    convTcn_kernel<<<dim3(192), blk, 0, stream>>>(tcn_w2 + (size_t)i * 49152, tcnT[2 * i + 1]);
  }

  for (int i = 0; i < 3; ++i) {
    int d = 1 << i;
    tcn_conv_kernel<<<dim3((unsigned)(BN / 8)), blk, 0, stream>>>(
        h16, tcnT[2 * i], tcn_b1 + i * 128, z16, nullptr, d);
    tcn_conv_kernel<<<dim3((unsigned)(BN / 8)), blk, 0, stream>>>(
        z16, tcnT[2 * i + 1], tcn_b2 + i * 128, h16, h16, d);
  }
  mask_kernel<<<dim3((unsigned)BN), blk, 0, stream>>>(h16, nbact);

  gemm_kernel<<<dim3(128, 8), blk, 0, stream>>>(h16, wencT, b_enc, feat, nullptr,
                                                16384, 512, 2048, 0);
  gemm_kernel<<<dim3(128, 8), blk, 0, stream>>>(feat, wqT, bq, qb, nullptr,
                                                16384, 512, 512, 0);
  gemm_kernel<<<dim3(128, 8), blk, 0, stream>>>(feat, wkT, bk, kb, nullptr,
                                                16384, 512, 512, 0);
  gemm_kernel<<<dim3(128, 8), blk, 0, stream>>>(feat, wvT, bv, vb, nullptr,
                                                16384, 512, 512, 0);
  attn_kernel<<<dim3(512), blk, 0, stream>>>(qb, kb, vb, ob);
  gemm_kernel<<<dim3(128, 8), blk, 0, stream>>>(ob, woT, bo, projb, nullptr,
                                                16384, 512, 512, 0);
  gemm_kernel<<<dim3(128, 16), blk, 0, stream>>>(projb, wp1T, b_p1, p1b, nullptr,
                                                 16384, 1024, 512, 1);
  gemm_kernel<<<dim3(128, 16), blk, 0, stream>>>(p1b, wp2T, b_p2, p2b, nullptr,
                                                 16384, 1024, 1024, 1);
  gemm_kernel<<<dim3(128, 1), blk, 0, stream>>>(p2b, wp3T, b_p3, nullptr,
                                                (float*)d_out, 16384, 24, 1024, 0);
}
